// EdgeAttention_16269336117304
// MI455X (gfx1250) — matrix operand order repaired and checked by reading, not compiled, not run on hardware
//
#include <hip/hip_runtime.h>
#include <hip/hip_bf16.h>
#include <math.h>

// Problem constants (from reference)
#define NN    2048
#define KZ    20
#define ROWS  (NN * KZ)        // 40960
#define IFZ   128
#define AHZ   4
#define AFZ   32
#define PCOLS 512              // q | k | v | gate, 128 cols each

typedef __attribute__((ext_vector_type(16))) __bf16 v16bf;
typedef __attribute__((ext_vector_type(8)))  __bf16 v8bf;
typedef __attribute__((ext_vector_type(8)))  float  v8f;

static __device__ __forceinline__ v8f wmma_bf16(v16bf a, v16bf b, v8f c) {
  // 8 args: (neg_a, A, neg_b, B, c_mod, C, reuse_a, reuse_b)
  return __builtin_amdgcn_wmma_f32_16x16x32_bf16(false, a, false, b, (short)0, c,
                                                 false, false);
}

// ---------------------------------------------------------------- prep kernels
__global__ void k_cvt_x(const float* __restrict__ x, __bf16* __restrict__ xb,
                        int total) {
  int i = blockIdx.x * blockDim.x + threadIdx.x;
  if (i < total) xb[i] = (__bf16)x[i];
}

// Build Wt[640][128] bf16: rows 0-511 = {Wq,Wk,Wv,Wg}^T, rows 512-639 = Wback^T
__global__ void k_pack_w(const float* __restrict__ Wq, const float* __restrict__ Wk,
                         const float* __restrict__ Wv, const float* __restrict__ Wg,
                         const float* __restrict__ Wback, __bf16* __restrict__ wt) {
  int i = blockIdx.x * blockDim.x + threadIdx.x;  // < 5*128*128
  if (i >= 5 * 128 * 128) return;
  int w = i >> 14;
  int r = i & 16383;
  int n = r >> 7;
  int k = r & 127;
  const float* src = (w == 0) ? Wq : (w == 1) ? Wk : (w == 2) ? Wv
                   : (w == 3) ? Wg : Wback;
  wt[i] = (__bf16)src[k * 128 + n];   // transpose: Wt[n][k] = W[k][n]
}

// b = x @ Wb  (128 -> 4), fp32
__global__ void k_bproj(const float* __restrict__ x, const float* __restrict__ Wb,
                        float* __restrict__ bbuf) {
  int i = blockIdx.x * blockDim.x + threadIdx.x;
  if (i >= ROWS * AHZ) return;
  int row = i >> 2, j = i & 3;
  const float* xr = x + (size_t)row * IFZ;
  float s = 0.f;
#pragma unroll 8
  for (int k = 0; k < IFZ; k++) s += xr[k] * Wb[k * AHZ + j];
  bbuf[i] = s;
}

// ------------------------------------------------------------ WMMA bf16 GEMM
// C[M][Ncols] = A[M][128] * Wt[Ncols][128]^T ; sigmoid(. + bg) on cols >= gateStart
__global__ void k_wmma_gemm(const __bf16* __restrict__ A, const __bf16* __restrict__ Wt,
                            float* __restrict__ C, int M, int Ncols,
                            int gateStart, const float* __restrict__ bg) {
  const int lane  = threadIdx.x & 31;
  const int wave  = blockIdx.x * (blockDim.x >> 5) + (threadIdx.x >> 5);
  const int tilesN = Ncols >> 6;          // 64-col jobs per row tile
  const int mTile = wave / tilesN;
  const int nJob  = wave % tilesN;
  const int m0 = mTile << 4;
  if (m0 >= M) return;
  const int n0 = nJob << 6;

  // WMMA A operand: lane holds row m0+(lane&15); K = e + 16*(lane>=16)
  const int aK = (lane >> 4) << 4;        // 0 or 16
  // WMMA B operand: lane holds col n0+(lane&15); K = e + 16*(lane>=16)
  const int bK = (lane >> 4) << 4;        // 0 or 16

  const __bf16* aRow  = A  + (size_t)(m0 + (lane & 15)) * IFZ;
  const __bf16* bCol0 = Wt + (size_t)(n0 + (lane & 15)) * IFZ;
  __builtin_prefetch(bCol0, 0, 1);

  v8f acc0 = {}, acc1 = {}, acc2 = {}, acc3 = {};
#pragma unroll
  for (int ks = 0; ks < IFZ; ks += 32) {
    v8bf alo = *(const v8bf*)(aRow + ks + aK);
    v8bf ahi = *(const v8bf*)(aRow + ks + aK + 8);
    v16bf a = __builtin_shufflevector(alo, ahi, 0, 1, 2, 3, 4, 5, 6, 7,
                                      8, 9, 10, 11, 12, 13, 14, 15);
    v16bf b0 = *(const v16bf*)(bCol0 + 0 * 16 * IFZ + ks + bK);
    v16bf b1 = *(const v16bf*)(bCol0 + 1 * 16 * IFZ + ks + bK);
    v16bf b2 = *(const v16bf*)(bCol0 + 2 * 16 * IFZ + ks + bK);
    v16bf b3 = *(const v16bf*)(bCol0 + 3 * 16 * IFZ + ks + bK);
    acc0 = wmma_bf16(a, b0, acc0);
    acc1 = wmma_bf16(a, b1, acc1);
    acc2 = wmma_bf16(a, b2, acc2);
    acc3 = wmma_bf16(a, b3, acc3);
  }

  // C/D layout: VGPR r -> row m0 + r + 8*(lane>=16), col n0 + t*16 + (lane&15)
  const int rB = m0 + ((lane >> 4) << 3);
#pragma unroll
  for (int t = 0; t < 4; t++) {
    v8f acc = (t == 0) ? acc0 : (t == 1) ? acc1 : (t == 2) ? acc2 : acc3;
    int col = n0 + t * 16 + (lane & 15);
    bool g = col >= gateStart;
    float bgv = g ? bg[col - gateStart] : 0.f;
#pragma unroll
    for (int r = 0; r < 8; r++) {
      float v = acc[r];
      if (g) v = 1.f / (1.f + __expf(-(v + bgv)));
      C[(size_t)(rB + r) * Ncols + col] = v;
    }
  }
}

// ---------------------------------------------------------------- attention
// One block per node; one thread per (query k, head a). 40-key softmax.
__global__ void k_attn(const float* __restrict__ proj, const float* __restrict__ bbuf,
                       const int* __restrict__ edge, __bf16* __restrict__ attnb) {
  const int n = blockIdx.x;
  const int t = threadIdx.x;
  if (t >= KZ * AHZ) return;
  const int a  = t & 3;
  const int kq = t >> 2;
  const size_t rowq = (size_t)n * KZ + kq;
  const float* qp = proj + rowq * PCOLS + a * AFZ;

  float q[AFZ];
#pragma unroll
  for (int c = 0; c < AFZ; c++) q[c] = qp[c];

  const float invScale = 0.17677669529663687f;  // 1/sqrt(AFZ)
  float s[2 * KZ];
#pragma unroll
  for (int j = 0; j < KZ; j++) {               // self keys
    const float* kp = proj + ((size_t)n * KZ + j) * PCOLS + IFZ + a * AFZ;
    float d = 0.f;
#pragma unroll
    for (int c = 0; c < AFZ; c++) d += q[c] * kp[c];
    s[j] = d * invScale + bbuf[((size_t)n * KZ + j) * AHZ + a];
  }
  const int e = edge[(size_t)n * KZ + kq];
#pragma unroll
  for (int j = 0; j < KZ; j++) {               // neighbor keys
    const float* kp = proj + ((size_t)e * KZ + j) * PCOLS + IFZ + a * AFZ;
    float d = 0.f;
#pragma unroll
    for (int c = 0; c < AFZ; c++) d += q[c] * kp[c];
    s[KZ + j] = d * invScale + bbuf[((size_t)e * KZ + j) * AHZ + a];
  }

  float mx = s[0];
#pragma unroll
  for (int j = 1; j < 2 * KZ; j++) mx = fmaxf(mx, s[j]);
  float sum = 0.f;
#pragma unroll
  for (int j = 0; j < 2 * KZ; j++) { s[j] = __expf(s[j] - mx); sum += s[j]; }
  const float inv = 1.f / sum;

  float ctx[AFZ];
#pragma unroll
  for (int c = 0; c < AFZ; c++) ctx[c] = 0.f;
#pragma unroll
  for (int j = 0; j < KZ; j++) {
    const float w = s[j] * inv;
    const float* vp = proj + ((size_t)n * KZ + j) * PCOLS + 2 * IFZ + a * AFZ;
#pragma unroll
    for (int c = 0; c < AFZ; c++) ctx[c] += w * vp[c];
  }
#pragma unroll
  for (int j = 0; j < KZ; j++) {
    const float w = s[KZ + j] * inv;
    const float* vp = proj + ((size_t)e * KZ + j) * PCOLS + 2 * IFZ + a * AFZ;
#pragma unroll
    for (int c = 0; c < AFZ; c++) ctx[c] += w * vp[c];
  }

  const float* gp = proj + rowq * PCOLS + 3 * IFZ + a * AFZ;  // sigmoid already applied
  __bf16* op = attnb + rowq * IFZ + a * AFZ;
#pragma unroll
  for (int c = 0; c < AFZ; c++) op[c] = (__bf16)(gp[c] * ctx[c]);
}

// ------------------------------------------------- residual + LayerNorm (wave/row)
__global__ void k_ln(const float* __restrict__ back, const float* __restrict__ x,
                     const float* __restrict__ bback, const float* __restrict__ gamma,
                     const float* __restrict__ beta, float* __restrict__ out,
                     int rows) {
  const int lane = threadIdx.x & 31;
  const int row  = blockIdx.x * (blockDim.x >> 5) + (threadIdx.x >> 5);
  if (row >= rows) return;
  const size_t base = (size_t)row * IFZ;
  const float sq2 = 1.4142135623730951f;
  float y[4];
  float sum = 0.f;
#pragma unroll
  for (int i = 0; i < 4; i++) {
    int c = i * 32 + lane;
    y[i] = back[base + c] + bback[c] + sq2 * x[base + c];
    sum += y[i];
  }
#pragma unroll
  for (int off = 16; off > 0; off >>= 1) sum += __shfl_xor(sum, off, 32);
  const float mu = sum * (1.f / IFZ);
  float vs = 0.f;
#pragma unroll
  for (int i = 0; i < 4; i++) { float d = y[i] - mu; vs += d * d; }
#pragma unroll
  for (int off = 16; off > 0; off >>= 1) vs += __shfl_xor(vs, off, 32);
  const float inv = rsqrtf(vs * (1.f / IFZ) + 1e-5f);
#pragma unroll
  for (int i = 0; i < 4; i++) {
    int c = i * 32 + lane;
    out[base + c] = gamma[c] * (y[i] - mu) * inv + beta[c];
  }
}

// ------------------------------------------------------------------- launcher
extern "C" void kernel_launch(void* const* d_in, const int* in_sizes, int n_in,
                              void* d_out, int out_size, void* d_ws, size_t ws_size,
                              hipStream_t stream) {
  (void)in_sizes; (void)n_in; (void)out_size; (void)ws_size;
  const float* x     = (const float*)d_in[0];
  const int*   edge  = (const int*)d_in[1];
  const float* Wq    = (const float*)d_in[2];
  const float* Wk    = (const float*)d_in[3];
  const float* Wv    = (const float*)d_in[4];
  const float* Wb    = (const float*)d_in[5];
  const float* Wg    = (const float*)d_in[6];
  const float* bg    = (const float*)d_in[7];
  const float* Wback = (const float*)d_in[8];
  const float* bback = (const float*)d_in[9];
  const float* gamma = (const float*)d_in[10];
  const float* beta  = (const float*)d_in[11];
  float* out = (float*)d_out;

  // Workspace layout (all offsets 256B aligned)
  char* w = (char*)d_ws;
  __bf16* xb    = (__bf16*)(w + 0);            // 40960*128 bf16   = 10,485,760 B
  float*  proj  = (float*)(w + 10485760);      // 40960*512 f32    = 83,886,080 B
  __bf16* wt    = (__bf16*)(w + 94371840);     // 640*128 bf16     =    163,840 B
  float*  bbuf  = (float*)(w + 94535680);      // 40960*4 f32      =    655,360 B
  __bf16* attnb = (__bf16*)(w + 95191040);     // 40960*128 bf16   = 10,485,760 B
  float*  back  = (float*)(w + 105676800);     // 40960*128 f32    = 20,971,520 B

  k_cvt_x<<<(ROWS * IFZ + 255) / 256, 256, 0, stream>>>(x, xb, ROWS * IFZ);
  k_pack_w<<<(5 * 128 * 128 + 255) / 256, 256, 0, stream>>>(Wq, Wk, Wv, Wg, Wback, wt);
  k_bproj<<<(ROWS * AHZ + 255) / 256, 256, 0, stream>>>(x, Wb, bbuf);

  // Projection GEMM: M=40960, N=512 -> (40960/16)*(512/64)=20480 wave jobs, 8 waves/block
  k_wmma_gemm<<<2560, 256, 0, stream>>>(xb, wt, proj, ROWS, PCOLS, 384, bg);

  k_attn<<<NN, 256, 0, stream>>>(proj, bbuf, edge, attnb);

  // Back GEMM: M=40960, N=128 -> 5120 wave jobs -> 640 blocks; no gate epilogue
  k_wmma_gemm<<<640, 256, 0, stream>>>(attnb, wt + 512 * 128, back, ROWS, IFZ,
                                       1 << 30, nullptr);

  k_ln<<<ROWS / 8, 256, 0, stream>>>(back, x, bback, gamma, beta, out, ROWS);
}
